// InternImageLayer_11931419148987
// MI455X (gfx1250) — matrix operand order repaired and checked by reading, not compiled, not run on hardware
//
#include <hip/hip_runtime.h>
#include <hip/hip_bf16.h>
#include <math.h>

// ---------------------------------------------------------------------------
// InternImage DCNv4 block for MI455X (gfx1250, wave32, WMMA).
// Pipeline: LN1(transpose-in) -> [vproj, om] WMMA GEMMs -> bilinear deform
// sampling -> oproj WMMA GEMM (+residual) -> LN2 -> fc1 WMMA GEMM (+GELU) ->
// fc2 WMMA GEMM (+residual, fused NHWC->NCHW transpose).
// All GEMMs: f16 operands, f32 accumulation via v_wmma_f32_16x16x32_f16.
// ---------------------------------------------------------------------------

typedef _Float16 f16;
typedef __attribute__((ext_vector_type(16))) _Float16 v16h;
typedef __attribute__((ext_vector_type(8)))  _Float16 v8h;
typedef __attribute__((ext_vector_type(8)))  float    v8f;

#define CDIM 256
#define GDIM 16
#define CG 16
#define KT 9
#define HWD 64
#define HWTOT 4096   // 64*64
#define MTOK 16384   // B * H * W
#define OMN 432      // G*3*K
#define MLPH 1024

// ---------------------------------------------------------------------------
// LayerNorm over C=256. TRANSPOSED_IN: read x as (B,C,H,W); else (M,C).
// Writes optional fp32 copy of the (M,C) input and the f16 normalized row.
// ---------------------------------------------------------------------------
template <bool TRANSPOSED_IN>
__global__ __launch_bounds__(256) void ln_kernel(const float* __restrict__ x,
                                                 const float* __restrict__ w,
                                                 const float* __restrict__ b,
                                                 float* __restrict__ copyOut,
                                                 f16* __restrict__ outh) {
  const int token = blockIdx.x;     // 0..MTOK-1
  const int c = threadIdx.x;        // 0..255
  float val;
  if (TRANSPOSED_IN) {
    const int bb = token >> 12;           // /4096
    const int hw = token & 4095;
    val = x[(((size_t)bb * CDIM) + c) * HWTOT + hw];
  } else {
    val = x[(size_t)token * CDIM + c];
  }
  __shared__ float sb[256];
  sb[c] = val;
  __syncthreads();
  for (int st = 128; st > 0; st >>= 1) {
    if (c < st) sb[c] += sb[c + st];
    __syncthreads();
  }
  const float mu = sb[0] * (1.0f / 256.0f);
  __syncthreads();
  const float d = val - mu;
  sb[c] = d * d;
  __syncthreads();
  for (int st = 128; st > 0; st >>= 1) {
    if (c < st) sb[c] += sb[c + st];
    __syncthreads();
  }
  const float var = sb[0] * (1.0f / 256.0f);
  const float xn = d * rsqrtf(var + 1e-5f) * w[c] + b[c];
  if (copyOut) copyOut[(size_t)token * CDIM + c] = val;
  outh[(size_t)token * CDIM + c] = (f16)xn;
}

// ---------------------------------------------------------------------------
// fp32 -> f16 weight conversion
// ---------------------------------------------------------------------------
__global__ __launch_bounds__(256) void cvt_f16_kernel(const float* __restrict__ s,
                                                      f16* __restrict__ d, int n) {
  int i = blockIdx.x * 256 + threadIdx.x;
  if (i < n) d[i] = (f16)s[i];
}

// ---------------------------------------------------------------------------
// WMMA GEMM: out[M,N] = A[M,K] @ Wt[N,K]^T + bias, fused epilogues.
// Block tile 128x64, 8 waves (wave32), each wave a 32x32 sub-tile = 2x2 WMMA.
// MODE 0: f32 out            MODE 1: f32 out + residual
// MODE 2: f16 out, exact GELU
// MODE 3: f32 out + residual, fused (M,N)->(B,C,H,W) transpose into d_out
// ---------------------------------------------------------------------------
#define LDA_STRIDE 40  // halfs; 80B padded row stride (no half-wave conflicts)

template <int MODE>
__global__ __launch_bounds__(256) void gemm_wmma_kernel(
    const f16* __restrict__ A, const f16* __restrict__ Wt,
    const float* __restrict__ bias, int N, int K,
    float* __restrict__ outF, f16* __restrict__ outH,
    const float* __restrict__ res) {
  __shared__ f16 lA[128 * LDA_STRIDE];
  __shared__ f16 lB[64 * LDA_STRIDE];

  const int tid = threadIdx.x;
  const int wid = tid >> 5;          // 0..7
  const int lane = tid & 31;
  const int l16 = lane & 15;
  const int hiSel = lane >> 4;       // 0: lanes 0-15, 1: lanes 16-31
  const int wm = (wid >> 1) * 32;    // 0,32,64,96
  const int wn = (wid & 1) * 32;     // 0,32
  const int mBase = blockIdx.x * 128;
  const int nBase = blockIdx.y * 64;

  v8f acc[2][2] = {};

  union V16 { v16h v; struct { v8h lo, hi; } p; };

  for (int k0 = 0; k0 < K; k0 += 32) {
    __syncthreads();
    // A tile: 128 rows x 32 halfs, 512 x 8-half chunks, 2 per thread.
#pragma unroll
    for (int i = 0; i < 2; ++i) {
      const int chunk = tid + i * 256;      // 0..511
      const int row = chunk >> 2;
      const int cc = (chunk & 3) * 8;
      *(v8h*)(lA + row * LDA_STRIDE + cc) =
          *(const v8h*)(A + (size_t)(mBase + row) * K + k0 + cc);
    }
    // B tile: 64 rows x 32 halfs, 256 chunks, 1 per thread (N edge guarded).
    {
      const int row = tid >> 2;
      const int cc = (tid & 3) * 8;
      v8h bv = {};
      const int n = nBase + row;
      if (n < N) bv = *(const v8h*)(Wt + (size_t)n * K + k0 + cc);
      *(v8h*)(lB + row * LDA_STRIDE + cc) = bv;
    }
    __syncthreads();

    V16 af[2], bf[2];
#pragma unroll
    for (int i = 0; i < 2; ++i) {   // A fragments (ISA 16-bit A 16x32 layout)
      const int row = wm + i * 16 + l16;
      const int c0 = hiSel * 8;
      af[i].p.lo = *(const v8h*)(lA + row * LDA_STRIDE + c0);
      af[i].p.hi = *(const v8h*)(lA + row * LDA_STRIDE + c0 + 16);
    }
#pragma unroll
    for (int j = 0; j < 2; ++j) {   // B fragments (K-contiguous weight rows)
      const int row = wn + j * 16 + l16;
      const int c0 = hiSel * 8;
      bf[j].p.lo = *(const v8h*)(lB + row * LDA_STRIDE + c0);
      bf[j].p.hi = *(const v8h*)(lB + row * LDA_STRIDE + c0 + 16);
    }
#pragma unroll
    for (int i = 0; i < 2; ++i)
#pragma unroll
      for (int j = 0; j < 2; ++j)
        acc[i][j] = __builtin_amdgcn_wmma_f32_16x16x32_f16(
            false, af[i].v, false, bf[j].v, (short)0, acc[i][j], false, false);
  }

  // Epilogue. D layout: lane n = l16, vgpr r -> m = r + 8*hiSel.
  const int mo = hiSel * 8;
#pragma unroll
  for (int i = 0; i < 2; ++i) {
#pragma unroll
    for (int j = 0; j < 2; ++j) {
      const int n = nBase + wn + j * 16 + l16;
      if (n >= N) continue;
      const float bv = bias[n];
#pragma unroll
      for (int r = 0; r < 8; ++r) {
        const int m = mBase + wm + i * 16 + mo + r;
        float val = acc[i][j][r] + bv;
        if (MODE == 0) {
          outF[(size_t)m * N + n] = val;
        } else if (MODE == 1) {
          outF[(size_t)m * N + n] = val + res[(size_t)m * N + n];
        } else if (MODE == 2) {
          const float g = 0.5f * val * (1.0f + erff(val * 0.70710678118654752f));
          outH[(size_t)m * N + n] = (f16)g;
        } else {  // MODE 3: residual + NHWC->NCHW transpose
          const float o = val + res[(size_t)m * N + n];
          const int bb = m >> 12;
          const int hw = m & 4095;
          outF[(((size_t)bb * CDIM + n) << 12) + hw] = o;
        }
      }
    }
  }
}

// ---------------------------------------------------------------------------
// DCNv4 bilinear sampling + mask reduction.
// One (token m, group g) item per 16 lanes; lane%16 = channel within group.
// ---------------------------------------------------------------------------
__global__ __launch_bounds__(256) void dcn_sample_kernel(
    const float* __restrict__ v,    // (M, 256)
    const float* __restrict__ om,   // (M, 432)
    f16* __restrict__ samp) {       // (M, 256) f16
  const int item = blockIdx.x * 16 + (threadIdx.x >> 4);  // 0..M*G-1
  const int c = threadIdx.x & 15;
  const int m = item >> 4;
  const int g = item & 15;
  const int bb = m >> 12;
  const int hw = m & 4095;
  const int xw = hw & 63;
  const int yh = hw >> 6;
  const float* omp = om + (size_t)m * OMN + g * 27;
  const size_t vbase = ((size_t)bb << 12) * CDIM + g * CG + c;

  float accv = 0.0f;
#pragma unroll
  for (int k = 0; k < KT; ++k) {
    const float dx = (float)(k % 3) - 1.0f;
    const float dy = (float)(k / 3) - 1.0f;
    const float px = (float)xw + dx + omp[2 * k];
    const float py = (float)yh + dy + omp[2 * k + 1];
    const float mk = omp[18 + k];
    const float x0f = floorf(px), y0f = floorf(py);
    const float tx = px - x0f, ty = py - y0f;
    const int x0 = (int)x0f, y0 = (int)y0f;

    float s00 = 0.f, s10 = 0.f, s01 = 0.f, s11 = 0.f;
    if (x0 >= 0 && x0 < HWD && y0 >= 0 && y0 < HWD)
      s00 = v[vbase + (size_t)(y0 * HWD + x0) * CDIM];
    if (x0 + 1 >= 0 && x0 + 1 < HWD && y0 >= 0 && y0 < HWD)
      s10 = v[vbase + (size_t)(y0 * HWD + x0 + 1) * CDIM];
    if (x0 >= 0 && x0 < HWD && y0 + 1 >= 0 && y0 + 1 < HWD)
      s01 = v[vbase + (size_t)((y0 + 1) * HWD + x0) * CDIM];
    if (x0 + 1 >= 0 && x0 + 1 < HWD && y0 + 1 >= 0 && y0 + 1 < HWD)
      s11 = v[vbase + (size_t)((y0 + 1) * HWD + x0 + 1) * CDIM];

    const float bi = s00 * (1.f - tx) * (1.f - ty) + s10 * tx * (1.f - ty) +
                     s01 * (1.f - tx) * ty + s11 * tx * ty;
    accv += bi * mk;
  }
  samp[(size_t)m * CDIM + g * CG + c] = (f16)accv;
}

// ---------------------------------------------------------------------------
// Host launcher
// ---------------------------------------------------------------------------
extern "C" void kernel_launch(void* const* d_in, const int* in_sizes, int n_in,
                              void* d_out, int out_size, void* d_ws, size_t ws_size,
                              hipStream_t stream) {
  (void)in_sizes; (void)n_in; (void)out_size; (void)ws_size;
  const float* x        = (const float*)d_in[0];
  const float* norm1_w  = (const float*)d_in[1];
  const float* norm1_b  = (const float*)d_in[2];
  const float* vproj_w  = (const float*)d_in[3];
  const float* vproj_b  = (const float*)d_in[4];
  const float* om_w     = (const float*)d_in[5];
  const float* om_b     = (const float*)d_in[6];
  const float* oproj_w  = (const float*)d_in[7];
  const float* oproj_b  = (const float*)d_in[8];
  const float* norm2_w  = (const float*)d_in[9];
  const float* norm2_b  = (const float*)d_in[10];
  const float* fc1_w    = (const float*)d_in[11];
  const float* fc1_b    = (const float*)d_in[12];
  const float* fc2_w    = (const float*)d_in[13];
  const float* fc2_b    = (const float*)d_in[14];

  char* ws = (char*)d_ws;
  size_t off = 0;
  auto alloc = [&](size_t bytes) -> char* {
    char* p = ws + off;
    off += (bytes + 255) & ~(size_t)255;
    return p;
  };
  float* shortcut = (float*)alloc((size_t)MTOK * CDIM * 4);
  f16*   xnh      = (f16*)  alloc((size_t)MTOK * CDIM * 2);
  float* vbuf     = (float*)alloc((size_t)MTOK * CDIM * 4);
  float* ombuf    = (float*)alloc((size_t)MTOK * OMN * 4);
  f16*   samph    = (f16*)  alloc((size_t)MTOK * CDIM * 2);
  float* x1       = (float*)alloc((size_t)MTOK * CDIM * 4);
  f16*   h2h      = (f16*)  alloc((size_t)MTOK * CDIM * 2);
  f16*   h1h      = (f16*)  alloc((size_t)MTOK * MLPH * 2);
  f16*   wv_h     = (f16*)  alloc((size_t)CDIM * CDIM * 2);
  f16*   wom_h    = (f16*)  alloc((size_t)OMN * CDIM * 2);
  f16*   wop_h    = (f16*)  alloc((size_t)CDIM * CDIM * 2);
  f16*   wf1_h    = (f16*)  alloc((size_t)MLPH * CDIM * 2);
  f16*   wf2_h    = (f16*)  alloc((size_t)CDIM * MLPH * 2);

  // 1) LN1 with NCHW->NLC transpose; keep fp32 shortcut, f16 normalized row.
  ln_kernel<true><<<MTOK, 256, 0, stream>>>(x, norm1_w, norm1_b, shortcut, xnh);

  // 2) Weight conversion fp32 -> f16.
  auto cvt = [&](const float* s, f16* d, int n) {
    cvt_f16_kernel<<<(n + 255) / 256, 256, 0, stream>>>(s, d, n);
  };
  cvt(vproj_w, wv_h, CDIM * CDIM);
  cvt(om_w, wom_h, OMN * CDIM);
  cvt(oproj_w, wop_h, CDIM * CDIM);
  cvt(fc1_w, wf1_h, MLPH * CDIM);
  cvt(fc2_w, wf2_h, CDIM * MLPH);

  const dim3 blk(256);
  // 3) v = xn @ vproj^T + b   (M x 256, K=256)
  gemm_wmma_kernel<0><<<dim3(MTOK / 128, CDIM / 64), blk, 0, stream>>>(
      xnh, wv_h, vproj_b, CDIM, CDIM, vbuf, nullptr, nullptr);
  // 4) om = xn @ om_w^T + b   (M x 432, K=256)
  gemm_wmma_kernel<0><<<dim3(MTOK / 128, (OMN + 63) / 64), blk, 0, stream>>>(
      xnh, wom_h, om_b, OMN, CDIM, ombuf, nullptr, nullptr);
  // 5) deformable bilinear sampling + mask -> f16 (M x 256)
  dcn_sample_kernel<<<(MTOK * GDIM) / 16, blk, 0, stream>>>(vbuf, ombuf, samph);
  // 6) x1 = shortcut + samp @ oproj^T + b
  gemm_wmma_kernel<1><<<dim3(MTOK / 128, CDIM / 64), blk, 0, stream>>>(
      samph, wop_h, oproj_b, CDIM, CDIM, x1, nullptr, shortcut);
  // 7) LN2 on x1 -> f16
  ln_kernel<false><<<MTOK, 256, 0, stream>>>(x1, norm2_w, norm2_b, nullptr, h2h);
  // 8) h1 = gelu(h2 @ fc1^T + b)  (M x 1024, K=256) -> f16
  gemm_wmma_kernel<2><<<dim3(MTOK / 128, MLPH / 64), blk, 0, stream>>>(
      h2h, wf1_h, fc1_b, MLPH, CDIM, nullptr, h1h, nullptr);
  // 9) out = transpose(x1 + h1 @ fc2^T + b)  (K=1024) -> d_out (B,C,H,W) f32
  gemm_wmma_kernel<3><<<dim3(MTOK / 128, CDIM / 64), blk, 0, stream>>>(
      h1h, wf2_h, fc2_b, CDIM, MLPH, (float*)d_out, nullptr, x1);
}
